// LSTMFixedLen_27943057228045
// MI455X (gfx1250) — compile-verified, matrix operand order checked by reading, not run on hardware
//
#include <hip/hip_runtime.h>
#include <hip/hip_bf16.h>
#include <stdint.h>

#define T_STEPS 14
#define EMB 16
#define HID 128
#define G4H 512
#define H_STRIDE 136   // bf16 row stride for h_sh: 68 dwords -> 4-bank row skew, conflict-free b128
#define E_STRIDE 232   // bf16 row stride for emb_sh (14*16 = 224, +8 pad)

typedef __attribute__((ext_vector_type(16))) __bf16 v16bf;
typedef __attribute__((ext_vector_type(8)))  float  v8f;

union AFrag { v16bf v; uint4 q[2]; };

__device__ __forceinline__ uint16_t f32_to_bf16(float f) {
  uint32_t u = __float_as_uint(f);
  u += 0x7FFFu + ((u >> 16) & 1u);           // round-to-nearest-even
  return (uint16_t)(u >> 16);
}

__device__ __forceinline__ float sigmoidf_fast(float x) {
  return 1.0f / (1.0f + __expf(-x));
}
__device__ __forceinline__ float tanhf_fast(float x) {
  float e = __expf(-2.0f * x);
  return (1.0f - e) / (1.0f + e);
}

// Packed-column p = nt*16+u maps to original W row r = gate*128 + group*16 + u,
// with nt = 4*group + gate  (gate order i,f,g,o per 16-unit group).
__device__ __forceinline__ int packed_row(int nt, int u) {
  return (nt & 3) * HID + (nt >> 2) * 16 + u;
}
// 16-bit operand fragment per-lane element order (mirrors ISA 16-bit A layout):
// elems 0..7 -> K = hi*8..hi*8+7 ; elems 8..15 -> K = 16+hi*8..16+hi*8+7
__device__ __forceinline__ int klocal(int hi, int e) {
  return (e < 8) ? (hi * 8 + e) : (16 + hi * 8 + (e - 8));
}

// ---------------- setup kernels: pack weights into WMMA fragment layout ----------------

__global__ void pack_whh_kernel(const float* __restrict__ W_hh, uint16_t* __restrict__ dst) {
  int tid = blockIdx.x * blockDim.x + threadIdx.x;     // 128 frags * 32 lanes * 16 elems = 65536
  int e = tid & 15, lane = (tid >> 4) & 31, frag = tid >> 9;
  int nt = frag >> 2, kc = frag & 3;
  int u = lane & 15, hi = lane >> 4;
  int r = packed_row(nt, u);
  int k = kc * 32 + klocal(hi, e);
  dst[frag * 512 + lane * 16 + e] = f32_to_bf16(W_hh[r * HID + k]);
}

__global__ void pack_wih_kernel(const float* __restrict__ W_ih, uint16_t* __restrict__ dst) {
  int tid = blockIdx.x * blockDim.x + threadIdx.x;     // 32 frags * 512 = 16384
  int e = tid & 15, lane = (tid >> 4) & 31, nt = tid >> 9;
  int u = lane & 15, hi = lane >> 4;
  int r = packed_row(nt, u);
  int kl = klocal(hi, e);
  float v = (kl < EMB) ? W_ih[r * EMB + kl] : 0.0f;    // K padded 16 -> 32 with zeros
  dst[nt * 512 + lane * 16 + e] = f32_to_bf16(v);
}

__global__ void pack_bias_kernel(const float* __restrict__ b_ih,
                                 const float* __restrict__ b_hh,
                                 float* __restrict__ dst) {
  int p = blockIdx.x * blockDim.x + threadIdx.x;
  if (p < G4H) {
    int nt = p >> 4, u = p & 15;
    int r = packed_row(nt, u);
    dst[p] = b_ih[r] + b_hh[r];
  }
}

// ---------------- fused persistent LSTM kernel ----------------
// Block = 256 threads = 8 waves; block processes 16-row batch tiles (grid-stride).
// Wave w owns gates i/f/g/o of hidden units [16w, 16w+16): 4 WMMA N-tiles whose
// D-layouts coincide -> elementwise LSTM update is register-local per lane.
// launch_bounds(256,4): force <=256 VGPRs/wave so 2 blocks co-reside per WGP.

__global__ void __launch_bounds__(256, 4)
lstm_fused_kernel(const int* __restrict__ x,
                  const float* __restrict__ emb_table,
                  const uint16_t* __restrict__ whh_pack,
                  const uint16_t* __restrict__ wih_pack,
                  const float* __restrict__ bias_pack,
                  const float* __restrict__ W1, const float* __restrict__ b1,
                  const float* __restrict__ W2, const float* __restrict__ b2,
                  const float* __restrict__ W3, const float* __restrict__ b3,
                  float* __restrict__ out, int numTiles, int B)
{
  __shared__ __align__(16) uint16_t h_sh[16 * H_STRIDE];
  __shared__ __align__(16) uint16_t emb_sh[16 * E_STRIDE];
  __shared__ __align__(16) uint16_t wih_sh[32 * 512];     // all 32 W_ih fragments (32 KiB)
  __shared__ float hf_sh[16 * HID];
  __shared__ float y1_sh[16 * 64];
  __shared__ float y2_sh[16 * 32];

  const int tid  = threadIdx.x;
  const int lane = tid & 31;
  const int w    = tid >> 5;     // wave 0..7
  const int col  = lane & 15;    // D-matrix column; also A-matrix row (M = lane&15)
  const int hi   = lane >> 4;

  // Stage W_ih fragments into LDS once per block (used once per step -> LDS, not VGPRs).
  {
    const uint4* src = (const uint4*)wih_pack;
    uint4* dst = (uint4*)wih_sh;
    for (int i = tid; i < (32 * 512) / 8; i += 256) dst[i] = src[i];
  }

  // Persistent per-wave W_hh B fragments (bf16) held in VGPRs for the entire kernel.
  v16bf whh_frag[4][4];          // [gate][k-chunk]  : 128 VGPRs
#pragma unroll
  for (int g = 0; g < 4; ++g) {
#pragma unroll
    for (int kc = 0; kc < 4; ++kc) {
      const uint4* p = (const uint4*)(whh_pack + (size_t)((4 * w + g) * 4 + kc) * 512 + lane * 16);
      AFrag f; f.q[0] = p[0]; f.q[1] = p[1];
      whh_frag[g][kc] = f.v;
    }
  }
  float biasv[4];
#pragma unroll
  for (int g = 0; g < 4; ++g) biasv[g] = bias_pack[64 * w + 16 * g + col];

  for (int tile = blockIdx.x; tile < numTiles; tile += gridDim.x) {
    const int rowBase = tile * 16;

    // Zero h, stage bf16 embeddings for all 14 steps of this tile.
    for (int i = tid; i < 16 * H_STRIDE; i += 256) h_sh[i] = 0;
    if (tid < 16 * T_STEPS) {
      int row = tid / T_STEPS, t = tid % T_STEPS;
      int gr = rowBase + row;
      int idx = (gr < B) ? x[gr * T_STEPS + t] : 0;
      const float* src = emb_table + (size_t)idx * EMB;
#pragma unroll
      for (int d = 0; d < EMB; ++d) {
        float v = (idx == 0) ? 0.0f : src[d];     // padding_idx = 0
        emb_sh[row * E_STRIDE + t * EMB + d] = f32_to_bf16(v);
      }
    }
    __syncthreads();

    v8f c;
#pragma unroll
    for (int r = 0; r < 8; ++r) c[r] = 0.0f;

    for (int t = 0; t < T_STEPS; ++t) {
      // Zero-C start: bias is added in the scalar epilogue instead of splatting
      // it into 32 accumulator VGPRs every step.
      v8f cz;
#pragma unroll
      for (int r = 0; r < 8; ++r) cz[r] = 0.0f;
      v8f acc[4];

      // x-projection: A = emb tile (K=16 real, padded to 32); B frags from LDS.
      AFrag ae;
      ae.q[0] = *(const uint4*)&emb_sh[col * E_STRIDE + t * EMB + hi * 8];
      uint4 z; z.x = z.y = z.z = z.w = 0u;
      ae.q[1] = z;
#pragma unroll
      for (int g = 0; g < 4; ++g) {
        AFrag bw;
        const uint4* p = (const uint4*)&wih_sh[(4 * w + g) * 512 + lane * 16];
        bw.q[0] = p[0]; bw.q[1] = p[1];
        acc[g] = __builtin_amdgcn_wmma_f32_16x16x32_bf16(
            false, ae.v, false, bw.v, (short)0, cz, false, false);
      }

      // recurrent GEMM: K = 128 over 4 chunks of 32, 16 WMMAs
#pragma unroll
      for (int kc = 0; kc < 4; ++kc) {
        AFrag ah;
        int base = col * H_STRIDE + kc * 32 + hi * 8;
        ah.q[0] = *(const uint4*)&h_sh[base];
        ah.q[1] = *(const uint4*)&h_sh[base + 16];
#pragma unroll
        for (int g = 0; g < 4; ++g)
          acc[g] = __builtin_amdgcn_wmma_f32_16x16x32_bf16(
              false, ah.v, false, whh_frag[g][kc], (short)0, acc[g], false, false);
      }
      __syncthreads();   // all waves finished reading h_sh

      // elementwise LSTM update — identical D layouts across the 4 gate tiles
#pragma unroll
      for (int r = 0; r < 8; ++r) {
        float iv = sigmoidf_fast(acc[0][r] + biasv[0]);
        float fv = sigmoidf_fast(acc[1][r] + biasv[1]);
        float gv = tanhf_fast(acc[2][r] + biasv[2]);
        float ov = sigmoidf_fast(acc[3][r] + biasv[3]);
        float cn = fv * c[r] + iv * gv;
        c[r] = cn;
        float hn = ov * tanhf_fast(cn);
        int row = r + 8 * hi;                    // D layout: M = vgpr + 8*(lane>>4)
        h_sh[row * H_STRIDE + 16 * w + col] = f32_to_bf16(hn);
        if (t == T_STEPS - 1) hf_sh[row * HID + 16 * w + col] = hn;
      }
      __syncthreads();   // h_new visible to all waves
    }

    // ---- MLP head (~1% of FLOPs): plain VALU dots over f32 h_last ----
    for (int idx = tid; idx < 16 * 64; idx += 256) {
      int row = idx >> 6, j = idx & 63;
      float s = b1[j];
      const float* hr = hf_sh + row * HID;
      const float* wr = W1 + j * HID;
#pragma unroll 8
      for (int k = 0; k < HID; ++k) s += hr[k] * wr[k];
      y1_sh[idx] = sigmoidf_fast(s);
    }
    __syncthreads();
    for (int idx = tid; idx < 16 * 32; idx += 256) {
      int row = idx >> 5, j = idx & 31;
      float s = b2[j];
      const float* yr = y1_sh + row * 64;
      const float* wr = W2 + j * 64;
#pragma unroll 8
      for (int k = 0; k < 64; ++k) s += yr[k] * wr[k];
      y2_sh[idx] = sigmoidf_fast(s);
    }
    __syncthreads();
    if (tid < 32) {
      int row = tid >> 1, j = tid & 1;
      int gr = rowBase + row;
      float s = b3[j];
      const float* yr = y2_sh + row * 32;
      const float* wr = W3 + j * 32;
#pragma unroll
      for (int k = 0; k < 32; ++k) s += yr[k] * wr[k];
      if (gr < B) out[gr * 2 + j] = s;
    }
    __syncthreads();   // before next tile reuses LDS
  }
}

// ---------------- host entry ----------------

extern "C" void kernel_launch(void* const* d_in, const int* in_sizes, int n_in,
                              void* d_out, int out_size, void* d_ws, size_t ws_size,
                              hipStream_t stream) {
  const int*   x     = (const int*)d_in[0];
  const float* table = (const float*)d_in[1];
  const float* W_ih  = (const float*)d_in[2];
  const float* W_hh  = (const float*)d_in[3];
  const float* b_ih  = (const float*)d_in[4];
  const float* b_hh  = (const float*)d_in[5];
  const float* W1    = (const float*)d_in[6];
  const float* b1    = (const float*)d_in[7];
  const float* W2    = (const float*)d_in[8];
  const float* b2    = (const float*)d_in[9];
  const float* W3    = (const float*)d_in[10];
  const float* b3    = (const float*)d_in[11];
  float* out = (float*)d_out;

  uint16_t* whh_pack  = (uint16_t*)d_ws;                              // 128 KiB
  uint16_t* wih_pack  = (uint16_t*)((char*)d_ws + 131072);            //  32 KiB
  float*    bias_pack = (float*)   ((char*)d_ws + 131072 + 32768);    //   2 KiB

  int B = in_sizes[0] / T_STEPS;
  int numTiles = (B + 15) / 16;

  pack_whh_kernel <<<256, 256, 0, stream>>>(W_hh, whh_pack);
  pack_wih_kernel <<<64,  256, 0, stream>>>(W_ih, wih_pack);
  pack_bias_kernel<<<2,   256, 0, stream>>>(b_ih, b_hh, bias_pack);

  int grid = numTiles < 2048 ? numTiles : 2048;
  lstm_fused_kernel<<<grid, 256, 0, stream>>>(x, table, whh_pack, wih_pack, bias_pack,
                                              W1, b1, W2, b2, W3, b3, out, numTiles, B);
}
